// GQA_82377472737314
// MI455X (gfx1250) — matrix operand order repaired and checked by reading, not compiled, not run on hardware
//
#include <hip/hip_runtime.h>
#include <math.h>

// ---- problem constants (from the reference) ----
#define Bq   2
#define Sq   1024
#define Dq   512
#define Gq   8
#define Hq   8
#define QKD  64
#define VD   64
#define NQ   (Gq * Hq * QKD)          // 4096
#define NKV  (Gq * (QKD + VD))        // 1024
#define MROWS (Bq * Sq)               // 2048
#define FFOUT 512

typedef __bf16 bf16;
typedef __attribute__((ext_vector_type(16))) bf16  v16bf;
typedef __attribute__((ext_vector_type(8)))  bf16  v8bf;
typedef __attribute__((ext_vector_type(8)))  float v8f;

// ---- helpers ----
static __device__ __forceinline__ bf16 f2bf(float f) {
  union { float f; unsigned u; } in; in.f = f;
  unsigned u = in.u;
  unsigned r = u + 0x7FFFu + ((u >> 16) & 1u);   // round-to-nearest-even
  union { unsigned short s; bf16 b; } out; out.s = (unsigned short)(r >> 16);
  return out.b;
}

// Build a 16-elem bf16 fragment from two 16B chunks at 32-bit element offsets
// off0/off1 from uniform base -> lowers to global_load_b128 saddr+voffset form.
static __device__ __forceinline__ v16bf ld16o(const bf16* __restrict__ base, int off0, int off1) {
  v8bf a = *(const v8bf*)(base + off0);
  v8bf b = *(const v8bf*)(base + off1);
  return __builtin_shufflevector(a, b, 0,1,2,3,4,5,6,7,8,9,10,11,12,13,14,15);
}

static __device__ __forceinline__ v16bf ld16p(const bf16* p0, const bf16* p1) {
  v8bf a = *(const v8bf*)p0;
  v8bf b = *(const v8bf*)p1;
  return __builtin_shufflevector(a, b, 0,1,2,3,4,5,6,7,8,9,10,11,12,13,14,15);
}

static __device__ __forceinline__ v8f wmma_bf(v16bf a, v16bf b, v8f c) {
  // v_wmma_f32_16x16x32_bf16  (neg_a, A, neg_b, B, c_mod, C, reuse_a, reuse_b)
  return __builtin_amdgcn_wmma_f32_16x16x32_bf16(false, a, false, b, (short)0, c, false, false);
}

static __device__ __forceinline__ float rmax16(float t) {  // reduce within 16-lane half
  t = fmaxf(t, __shfl_xor(t, 1, 32));
  t = fmaxf(t, __shfl_xor(t, 2, 32));
  t = fmaxf(t, __shfl_xor(t, 4, 32));
  t = fmaxf(t, __shfl_xor(t, 8, 32));
  return t;
}
static __device__ __forceinline__ float rsum16(float t) {
  t += __shfl_xor(t, 1, 32);
  t += __shfl_xor(t, 2, 32);
  t += __shfl_xor(t, 4, 32);
  t += __shfl_xor(t, 8, 32);
  return t;
}

// ---- conversion kernels ----
__global__ void cvt_f32_bf16(const float* __restrict__ in, bf16* __restrict__ out, int n) {
  int i = blockIdx.x * blockDim.x + threadIdx.x;
  if (i < n) out[i] = f2bf(in[i]);
}

// in: row-major [K,N] f32 -> out: row-major [N,K] bf16  (weight transpose+convert)
__global__ void tconv_kernel(const float* __restrict__ in, bf16* __restrict__ out, int K, int N) {
  int t = blockIdx.x * blockDim.x + threadIdx.x;
  if (t < K * N) {
    int n = t / K, k = t - n * K;
    out[t] = f2bf(in[(size_t)k * N + n]);
  }
}

// ---- generic bf16 WMMA GEMM: C[M,N] = A[M,K] @ Bt[N,K]^T + bias ----
// One wave computes a 64x64 tile (acc[4][4]); B-fragments are streamed to cap
// register pressure; all global addressing is 32-bit offsets from uniform bases.
// mode 0: out_bf[M,N] = bf16(C * 0.125)                  (q path, pre-scaled)
// mode 1: cols<512 -> kb[M,512]; cols>=512 -> vt[B,G,64,S] transposed  (kv path)
// mode 2: out_f[M,N] = leaky_relu(C)                     (final projection)
__global__ void __launch_bounds__(128, 1)
gemm_wmma(const bf16* __restrict__ A, const bf16* __restrict__ Bt,
          const float* __restrict__ bias, int M, int N, int K, int mode,
          bf16* __restrict__ out_bf, float* __restrict__ out_f,
          bf16* __restrict__ kb, bf16* __restrict__ vt) {
  const int lane = threadIdx.x & 31;
  const int w    = threadIdx.x >> 5;
  const int col16 = lane & 15;
  const int half  = lane >> 4;
  const int n0 = (blockIdx.x * 4 + w) * 64;
  const int m0 = blockIdx.y * 64;

  v8f acc[4][4] = {};

  // 32-bit element offsets (max < 2^25 elements -> fits easily)
  int aoff[4], boff[4];
#pragma unroll
  for (int i = 0; i < 4; ++i) aoff[i] = (m0 + 16 * i + col16) * K + half * 16;
#pragma unroll
  for (int j = 0; j < 4; ++j) boff[j] = (n0 + 16 * j + col16) * K + half * 16;

  for (int k0 = 0; k0 < K; k0 += 32) {
    v16bf af[4];
#pragma unroll
    for (int i = 0; i < 4; ++i) {               // A frag: ks {half*16 .. half*16+15}
      af[i] = ld16o(A, aoff[i], aoff[i] + 8);
      aoff[i] += 32;
    }
#pragma unroll
    for (int j = 0; j < 4; ++j) {               // stream B frags: only one live at a time
      v16bf bfm = ld16o(Bt, boff[j], boff[j] + 8);
      boff[j] += 32;
#pragma unroll
      for (int i = 0; i < 4; ++i)
        acc[i][j] = wmma_bf(af[i], bfm, acc[i][j]);
    }
  }

  float bcol[4];
#pragma unroll
  for (int j = 0; j < 4; ++j) bcol[j] = bias[n0 + 16 * j + col16];

#pragma unroll
  for (int i = 0; i < 4; ++i) {
#pragma unroll
    for (int j = 0; j < 4; ++j) {
#pragma unroll
      for (int r = 0; r < 8; ++r) {
        int row = m0 + 16 * i + r + 8 * half;   // C layout: VGPR r, lane-half selects M or M+8
        int col = n0 + 16 * j + col16;
        float v = acc[i][j][r] + bcol[j];
        if (mode == 0) {
          out_bf[(size_t)row * N + col] = f2bf(v * 0.125f);      // fold softmax scale (exact pow2)
        } else if (mode == 1) {
          if (col < Gq * QKD) {
            kb[(size_t)row * (Gq * QKD) + col] = f2bf(v);
          } else {
            int c = col - Gq * QKD;
            int g = c >> 6, d = c & 63;
            int b = row >> 10, s = row & 1023;
            vt[(((size_t)(b * Gq + g) * 64 + d) << 10) + s] = f2bf(v);  // V transposed
          }
        } else {
          v = v > 0.0f ? v : 0.01f * v;                          // leaky_relu
          out_f[(size_t)row * N + col] = v;
        }
      }
    }
  }
}

// ---- flash-attention per head: one wave = 16 query rows of one (b,g,h) ----
__global__ void __launch_bounds__(128, 1)
attn_kernel(const bf16* __restrict__ qb, const bf16* __restrict__ kb,
            const bf16* __restrict__ vt, bf16* __restrict__ zb) {
  __shared__ __attribute__((aligned(16))) bf16 pstage[4][16][32];  // per-wave P staging (C->A relayout)

  const int lane = threadIdx.x & 31;
  const int w    = threadIdx.x >> 5;
  const int col16 = lane & 15;
  const int half  = lane >> 4;

  const int head = blockIdx.y;              // b*64 + g*8 + h
  const int b = head >> 6;
  const int g = (head >> 3) & 7;
  const int h = head & 7;
  const int s0 = (blockIdx.x * 4 + w) * 16;
  const int qcol0 = (g * Hq + h) * QKD;

  // Q fragments (queries pre-scaled by 0.125 in the q-GEMM epilogue)
  v16bf qf[2];
  {
    int qoff = (b * Sq + s0 + col16) * NQ + qcol0 + half * 16;
    qf[0] = ld16o(qb, qoff,      qoff + 8);
    qf[1] = ld16o(qb, qoff + 32, qoff + 40);
  }

  v8f accv[4] = {};
  float m_run[8], l_run[8];
#pragma unroll
  for (int r = 0; r < 8; ++r) { m_run[r] = -INFINITY; l_run[r] = 0.0f; }

  // 32-bit element offsets into kb and vt, advanced per key tile
  // K frag (nt,ks): key = key0+nt*16+col16, row stride 512, ks*32 + half*16
  int koff0 = (b * Sq + col16) * (Gq * QKD) + g * QKD + half * 16;   // nt=0, ks=0
  // V frag (n): d = n*16+col16 -> vt row (d)*Sq, start key0 + half*16
  int voff0 = (b * Gq + g) * 64 * Sq + col16 * Sq + half * 16;

  for (int key0 = 0; key0 < Sq; key0 += 32) {
    // ---- S = Q @ K^T (two 16-key N-tiles, two 32-d k-steps) ----
    v8f c0 = {}, c1 = {};
    {
      int kbase0 = koff0 + key0 * (Gq * QKD);
      v16bf kf00 = ld16o(kb, kbase0,               kbase0 + 8);
      v16bf kf01 = ld16o(kb, kbase0 + 32,          kbase0 + 40);
      int kbase1 = kbase0 + 16 * (Gq * QKD);
      v16bf kf10 = ld16o(kb, kbase1,               kbase1 + 8);
      v16bf kf11 = ld16o(kb, kbase1 + 32,          kbase1 + 40);
      c0 = wmma_bf(qf[0], kf00, c0);  c0 = wmma_bf(qf[1], kf01, c0);
      c1 = wmma_bf(qf[0], kf10, c1);  c1 = wmma_bf(qf[1], kf11, c1);
    }

    // ---- online softmax (rows live within a 16-lane half per VGPR index) ----
#pragma unroll
    for (int r = 0; r < 8; ++r) {
      float t  = rmax16(fmaxf(c0[r], c1[r]));
      float nm = fmaxf(m_run[r], t);
      float al = __expf(m_run[r] - nm);
      float e0 = __expf(c0[r] - nm);
      float e1 = __expf(c1[r] - nm);
      float rs = rsum16(e0 + e1);
      l_run[r] = l_run[r] * al + rs;
      m_run[r] = nm;
#pragma unroll
      for (int n = 0; n < 4; ++n) accv[n][r] *= al;
      int prow = r + 8 * half;
      pstage[w][prow][col16]      = f2bf(e0);
      pstage[w][prow][16 + col16] = f2bf(e1);
    }
    asm volatile("s_wait_dscnt 0" ::: "memory");   // wave-local DS RAW: stores before A-frag reads

    // ---- acc += P @ V ----
    v16bf pf;
    {
      const bf16* pp = &pstage[w][col16][half * 16];
      pf = ld16p(pp, pp + 8);
    }
#pragma unroll
    for (int n = 0; n < 4; ++n) {
      int vo = voff0 + n * 16 * Sq + key0;   // V stored transposed -> contiguous along keys
      v16bf vf = ld16o(vt, vo, vo + 8);
      accv[n] = wmma_bf(pf, vf, accv[n]);
    }
  }

  // ---- normalize and emit z (bf16, [B,S, (g*H+h)*64 + d]) ----
#pragma unroll
  for (int r = 0; r < 8; ++r) {
    float inv = 1.0f / l_run[r];
    int row = b * Sq + s0 + r + 8 * half;
#pragma unroll
    for (int n = 0; n < 4; ++n)
      zb[(size_t)row * NQ + qcol0 + n * 16 + col16] = f2bf(accv[n][r] * inv);
  }
}

extern "C" void kernel_launch(void* const* d_in, const int* in_sizes, int n_in,
                              void* d_out, int out_size, void* d_ws, size_t ws_size,
                              hipStream_t stream) {
  const float* x     = (const float*)d_in[0];
  const float* Wq_w  = (const float*)d_in[1];
  const float* Wq_b  = (const float*)d_in[2];
  const float* Wkv_w = (const float*)d_in[3];
  const float* Wkv_b = (const float*)d_in[4];
  const float* Wz_w  = (const float*)d_in[5];
  const float* Wz_b  = (const float*)d_in[6];
  float* out = (float*)d_out;

  // workspace carve-up (bf16 elements); all sizes are multiples of 4KB -> aligned
  char* base = (char*)d_ws;
  size_t off = 0;
  auto take = [&](size_t elems) { bf16* p = (bf16*)(base + off); off += elems * sizeof(bf16); return p; };
  bf16* xb   = take((size_t)MROWS * Dq);        // 2048 x 512
  bf16* wqT  = take((size_t)NQ * Dq);           // 4096 x 512   (W^T)
  bf16* wkvT = take((size_t)NKV * Dq);          // 1024 x 512
  bf16* wzT  = take((size_t)FFOUT * NQ);        // 512 x 4096
  bf16* qbuf = take((size_t)MROWS * NQ);        // 2048 x 4096  (pre-scaled q)
  bf16* kbuf = take((size_t)MROWS * Gq * QKD);  // 2048 x 512
  bf16* vtb  = take((size_t)Bq * Gq * 64 * Sq); // V transposed: [B,G,64,S]
  bf16* zbuf = take((size_t)MROWS * NQ);        // 2048 x 4096

  // 0) convert activations / transpose-convert weights to bf16
  {
    int n = MROWS * Dq;
    cvt_f32_bf16<<<(n + 255) / 256, 256, 0, stream>>>(x, xb, n);
  }
  tconv_kernel<<<((Dq * NQ) + 255) / 256, 256, 0, stream>>>(Wq_w,  wqT,  Dq, NQ);
  tconv_kernel<<<((Dq * NKV) + 255) / 256, 256, 0, stream>>>(Wkv_w, wkvT, Dq, NKV);
  tconv_kernel<<<((NQ * FFOUT) + 255) / 256, 256, 0, stream>>>(Wz_w, wzT, NQ, FFOUT);

  // 1) q = (x @ Wq + b) * 0.125   -> qbuf
  gemm_wmma<<<dim3(NQ / 256, MROWS / 64), 128, 0, stream>>>(
      xb, wqT, Wq_b, MROWS, NQ, Dq, 0, qbuf, nullptr, nullptr, nullptr);

  // 2) kv = x @ Wkv + b  -> kbuf (K natural) + vtb (V transposed)
  gemm_wmma<<<dim3(NKV / 256, MROWS / 64), 128, 0, stream>>>(
      xb, wkvT, Wkv_b, MROWS, NKV, Dq, 1, nullptr, nullptr, kbuf, vtb);

  // 3) flash attention -> zbuf
  attn_kernel<<<dim3(Sq / 64, Bq * Gq * Hq), 128, 0, stream>>>(qbuf, kbuf, vtb, zbuf);

  // 4) out = leaky_relu(z @ Wz + b)  (f32)
  gemm_wmma<<<dim3(FFOUT / 256, MROWS / 64), 128, 0, stream>>>(
      zbuf, wzT, Wz_b, MROWS, FFOUT, NQ, 2, nullptr, out, nullptr, nullptr);
}
